// MBDSEvolved_70866960384241
// MI455X (gfx1250) — matrix operand order repaired and checked by reading, not compiled, not run on hardware
//
#include <hip/hip_runtime.h>
#include <hip/hip_bf16.h>

// ---------------------------------------------------------------------------
// Mamba diffusion forward for MI455X (gfx1250, wave32, WMMA bf16)
// B=4, S=2048, D=1024, DI=2048, DS=64, DR=64, DC=4
// ---------------------------------------------------------------------------

#define BB 4
#define SS 2048
#define DD 1024
#define DI 2048
#define DS_ 64
#define DR_ 64

typedef __attribute__((ext_vector_type(16))) __bf16 v16bf;
typedef __attribute__((ext_vector_type(8)))  float  v8f;

__device__ __forceinline__ float silu_f(float x) {
    return x * (1.0f / (1.0f + __expf(-x)));
}

// ---------------------------------------------------------------------------
// LDS-staged, double-buffered WMMA bf16 GEMM:
//   C[M,N] = epilogue( A[M,K] * W[N,K]^T + bias )
// A fp32 row-major (lda), W fp32 row-major (ldw), C fp32 row-major (ldc).
// Block = 256 threads (8 waves). Block tile 128M x BN, K-step 32.
// Wave grid 4(M) x 2(N); each wave owns 32M x (BN/2)N = 2 x (BN/32) WMMA accums.
// BN=128: branch-free loaders (16 floats/thread for both A and B), 8 WMMAs
// per wave per K-step. BN=64: for N=192 (x_proj).
// Global->regs prefetch of next K-slab overlaps WMMA on current LDS buffer.
// LDS rows padded to 40 bf16 (80B) to break 4-way bank conflicts.
// EPI: 0 = none, 1 = +bias, 2 = softplus(x + bias)
// ---------------------------------------------------------------------------
#define LDSPAD 40

template <int EPI, int BN>
__global__ __launch_bounds__(256) void gemm_wmma_bf16(
    const float* __restrict__ A, int lda,
    const float* __restrict__ W, int ldw,
    const float* __restrict__ bias,
    float* __restrict__ C, int ldc,
    int K)
{
    constexpr int NT = BN / 32;             // N-subtiles per wave (2 or 4)
    constexpr int WN = BN / 2;              // N strip per wave (32 or 64)

    __shared__ __align__(16) __bf16 shA[2][128][LDSPAD];
    __shared__ __align__(16) __bf16 shB[2][BN][LDSPAD];

    const int tid   = threadIdx.x;
    const int lane  = tid & 31;
    const int wave  = tid >> 5;
    const int waveM = wave >> 1;            // 0..3  -> 32-row strip
    const int waveN = wave & 1;             // 0..1  -> WN-col strip
    const int row0  = blockIdx.y * 128;
    const int col0  = blockIdx.x * BN;

    // global->LDS loader mapping: each thread owns 16 floats of A; for BN=128
    // each thread also owns 16 floats of B (branch-free); for BN=64 only
    // threads 0..127 load B.
    const int ldRow  = tid >> 1;            // 0..127
    const int ldHalf = tid & 1;             // k-half (16 floats)

    // fragment lane mapping (ISA 16-bit A 16x32 / B 32x16 layouts)
    const int mrow  = lane & 15;
    const int khalf = lane >> 4;
    const int ncol  = lane & 15;
    const int mofs  = khalf * 8;

    v8f acc[2][NT];
    #pragma unroll
    for (int mt = 0; mt < 2; ++mt)
        #pragma unroll
        for (int nt = 0; nt < NT; ++nt)
            #pragma unroll
            for (int r = 0; r < 8; ++r) acc[mt][nt][r] = 0.0f;

    float aP[16];
    float bP[16];

    const int nsteps = K / 32;

    auto gload = [&](int kb) {
        const float4* ga = (const float4*)(A + (size_t)(row0 + ldRow) * lda
                                             + kb + ldHalf * 16);
        float4 a0 = ga[0], a1 = ga[1], a2 = ga[2], a3 = ga[3];
        aP[0]=a0.x; aP[1]=a0.y; aP[2]=a0.z; aP[3]=a0.w;
        aP[4]=a1.x; aP[5]=a1.y; aP[6]=a1.z; aP[7]=a1.w;
        aP[8]=a2.x; aP[9]=a2.y; aP[10]=a2.z; aP[11]=a2.w;
        aP[12]=a3.x; aP[13]=a3.y; aP[14]=a3.z; aP[15]=a3.w;
        if constexpr (BN == 128) {
            const float4* gb = (const float4*)(W + (size_t)(col0 + ldRow) * ldw
                                                 + kb + ldHalf * 16);
            float4 b0 = gb[0], b1 = gb[1], b2 = gb[2], b3 = gb[3];
            bP[0]=b0.x; bP[1]=b0.y; bP[2]=b0.z; bP[3]=b0.w;
            bP[4]=b1.x; bP[5]=b1.y; bP[6]=b1.z; bP[7]=b1.w;
            bP[8]=b2.x; bP[9]=b2.y; bP[10]=b2.z; bP[11]=b2.w;
            bP[12]=b3.x; bP[13]=b3.y; bP[14]=b3.z; bP[15]=b3.w;
        } else {
            if (tid < 128) {
                const float4* gb = (const float4*)(W + (size_t)(col0 + ldRow) * ldw
                                                     + kb + ldHalf * 16);
                float4 b0 = gb[0], b1 = gb[1], b2 = gb[2], b3 = gb[3];
                bP[0]=b0.x; bP[1]=b0.y; bP[2]=b0.z; bP[3]=b0.w;
                bP[4]=b1.x; bP[5]=b1.y; bP[6]=b1.z; bP[7]=b1.w;
                bP[8]=b2.x; bP[9]=b2.y; bP[10]=b2.z; bP[11]=b2.w;
                bP[12]=b3.x; bP[13]=b3.y; bP[14]=b3.z; bP[15]=b3.w;
            }
        }
    };

    auto sstore = [&](int buf) {
        __bf16* da = &shA[buf][ldRow][ldHalf * 16];
        #pragma unroll
        for (int i = 0; i < 16; ++i) da[i] = (__bf16)aP[i];
        if constexpr (BN == 128) {
            __bf16* db = &shB[buf][ldRow][ldHalf * 16];
            #pragma unroll
            for (int i = 0; i < 16; ++i) db[i] = (__bf16)bP[i];
        } else {
            if (tid < 128) {
                __bf16* db = &shB[buf][ldRow][ldHalf * 16];
                #pragma unroll
                for (int i = 0; i < 16; ++i) db[i] = (__bf16)bP[i];
            }
        }
    };

    gload(0);
    sstore(0);
    __syncthreads();

    for (int step = 0; step < nsteps; ++step) {
        const int cur = step & 1;

        // prefetch next K-slab into registers (overlaps with WMMA below)
        if (step + 1 < nsteps) gload((step + 1) * 32);

        // ---- build fragments from LDS
        v16bf afr[2];
        #pragma unroll
        for (int mt = 0; mt < 2; ++mt) {
            const __bf16* pa = &shA[cur][waveM * 32 + mt * 16 + mrow][0];
            #pragma unroll
            for (int i = 0; i < 8; ++i) afr[mt][i]     = pa[khalf * 8 + i];
            #pragma unroll
            for (int i = 0; i < 8; ++i) afr[mt][8 + i] = pa[16 + khalf * 8 + i];
        }
        v16bf bfr[NT];
        #pragma unroll
        for (int nt = 0; nt < NT; ++nt) {
            const __bf16* pb = &shB[cur][waveN * WN + nt * 16 + ncol][0];
            #pragma unroll
            for (int i = 0; i < 8; ++i) bfr[nt][i]     = pb[khalf * 8 + i];
            #pragma unroll
            for (int i = 0; i < 8; ++i) bfr[nt][8 + i] = pb[16 + khalf * 8 + i];
        }

        #pragma unroll
        for (int mt = 0; mt < 2; ++mt)
            #pragma unroll
            for (int nt = 0; nt < NT; ++nt)
                acc[mt][nt] = __builtin_amdgcn_wmma_f32_16x16x32_bf16(
                                  false, afr[mt], false, bfr[nt],
                                  (short)0, acc[mt][nt], false, false);

        __syncthreads();                       // everyone done reading cur
        if (step + 1 < nsteps) sstore(cur ^ 1);
        __syncthreads();                       // next buffer visible
    }

    // ---- epilogue + store (lanes 0-15: M=r, lanes 16-31: M=r+8)
    #pragma unroll
    for (int mt = 0; mt < 2; ++mt) {
        #pragma unroll
        for (int nt = 0; nt < NT; ++nt) {
            const int col = col0 + waveN * WN + nt * 16 + ncol;
            float bv = (EPI >= 1) ? bias[col] : 0.0f;
            #pragma unroll
            for (int r = 0; r < 8; ++r) {
                float v = acc[mt][nt][r];
                if (EPI >= 1) v += bv;
                if (EPI == 2) v = (v > 20.0f) ? v : __logf(1.0f + __expf(v));
                C[(size_t)(row0 + waveM * 32 + mt * 16 + mofs + r) * ldc + col] = v;
            }
        }
    }
}

// ---------------------------------------------------------------------------
// t_proj[b,j] = sum_k emb(t_b)[k] * time_W[j,k] + time_b[j]   (B*D threads)
// ---------------------------------------------------------------------------
__global__ __launch_bounds__(256) void time_proj_kernel(
    const int* __restrict__ t, const float* __restrict__ time_W,
    const float* __restrict__ time_b, float* __restrict__ tproj)
{
    const int i = blockIdx.x * 256 + threadIdx.x;  // [0, B*D)
    const int b = i >> 10;
    const int j = i & 1023;
    const float tv = (float)t[b];
    const float c = -9.210340371976184f / 512.0f;  // -ln(10000)/half
    float acc = time_b[j];
    const float* wrow = time_W + (size_t)j * DD;
    for (int k = 0; k < DD; ++k) {
        float emb;
        if (k < 512) emb = __cosf(tv * __expf(c * (float)k));
        else         emb = __sinf(tv * __expf(c * (float)(k - 512)));
        acc += wrow[k] * emb;
    }
    tproj[i] = acc;
}

// ---------------------------------------------------------------------------
// h = x + t_proj[b,:] + PE[s,:]
// ---------------------------------------------------------------------------
__global__ __launch_bounds__(256) void build_h_kernel(
    const float* __restrict__ x, const float* __restrict__ tproj,
    float* __restrict__ h)
{
    const int i  = blockIdx.x * 256 + threadIdx.x;  // [0, B*S*D)
    const int dd = i & 1023;
    const int s  = (i >> 10) & 2047;
    const int b  = i >> 21;
    const int j2 = dd & ~1;
    const float div = __expf((float)j2 * (-9.210340371976184f / (float)DD));
    const float arg = (float)s * div;
    const float pe  = (dd & 1) ? __cosf(arg) : __sinf(arg);
    h[i] = x[i] + tproj[b * DD + dd] + pe;
}

// ---------------------------------------------------------------------------
// depthwise causal conv1d (DC=4) + SiLU over xz[:, :, 0:DI] -> xm
// ---------------------------------------------------------------------------
__global__ __launch_bounds__(256) void conv_silu_kernel(
    const float* __restrict__ xz, const float* __restrict__ cw,
    const float* __restrict__ cb, float* __restrict__ xm)
{
    const int i = blockIdx.x * 256 + threadIdx.x;   // [0, B*S*DI)
    const int c = i & 2047;
    const int s = (i >> 11) & 2047;
    const int b = i >> 22;
    const float w0 = cw[c * 4 + 0], w1 = cw[c * 4 + 1];
    const float w2 = cw[c * 4 + 2], w3 = cw[c * 4 + 3];
    const size_t base = ((size_t)b * SS + s) * (2 * DI) + c;
    float acc = cb[c] + w3 * xz[base];
    if (s >= 1) acc += w2 * xz[base - 1 * (2 * DI)];
    if (s >= 2) acc += w1 * xz[base - 2 * (2 * DI)];
    if (s >= 3) acc += w0 * xz[base - 3 * (2 * DI)];
    xm[((size_t)b * SS + s) * DI + c] = silu_f(acc);
}

// ---------------------------------------------------------------------------
// Selective scan. One thread per channel d; 64-element SSM state + A in regs.
// B_t, C_t staged in LDS per step. Fused epilogue:
//   xm[b,s,d] = (y + D_skip[d]*u) * silu(z)     (in-place over xm)
// ---------------------------------------------------------------------------
__global__ __launch_bounds__(256) void scan_kernel(
    const float* __restrict__ xdbl, const float* __restrict__ dtb,
    const float* __restrict__ xz,   const float* __restrict__ A_log,
    const float* __restrict__ Dskip, float* __restrict__ xm)
{
    const int b = blockIdx.x >> 3;                        // DI/256 = 8 chunks
    const int d = (blockIdx.x & 7) * 256 + threadIdx.x;

    float Areg[DS_], st[DS_];
    #pragma unroll
    for (int n = 0; n < DS_; ++n) {
        Areg[n] = -__expf(A_log[(size_t)d * DS_ + n]);
        st[n] = 0.0f;
    }
    const float Dd = Dskip[d];

    __shared__ float Bsh[DS_];
    __shared__ float Csh[DS_];

    for (int s = 0; s < SS; ++s) {
        const size_t row = (size_t)b * SS + s;
        if (threadIdx.x < 64)
            Bsh[threadIdx.x] = xdbl[row * 192 + 64 + threadIdx.x];
        else if (threadIdx.x < 128)
            Csh[threadIdx.x - 64] = xdbl[row * 192 + 128 + (threadIdx.x - 64)];
        __syncthreads();

        const float dt = dtb[row * DI + d];
        const float u  = xm[row * DI + d];
        const float z  = xz[row * (2 * DI) + DI + d];
        const float du = dt * u;
        float y = 0.0f;
        #pragma unroll
        for (int n = 0; n < DS_; ++n) {
            const float dA = __expf(dt * Areg[n]);
            const float v  = dA * st[n] + du * Bsh[n];
            st[n] = v;
            y += v * Csh[n];
        }
        xm[row * DI + d] = (y + Dd * u) * silu_f(z);
        __syncthreads();
    }
}

// ---------------------------------------------------------------------------
// In-place LayerNorm over rows of 1024 (one block per row).
// ---------------------------------------------------------------------------
__global__ __launch_bounds__(256) void layernorm_kernel(
    float* __restrict__ h, const float* __restrict__ g,
    const float* __restrict__ bta)
{
    __shared__ float red[256];
    const int tid = threadIdx.x;
    float* row = h + (size_t)blockIdx.x * DD;

    float v[4];
    float s = 0.0f;
    #pragma unroll
    for (int i = 0; i < 4; ++i) { v[i] = row[tid + 256 * i]; s += v[i]; }
    red[tid] = s; __syncthreads();
    for (int off = 128; off > 0; off >>= 1) {
        if (tid < off) red[tid] += red[tid + off];
        __syncthreads();
    }
    const float mu = red[0] * (1.0f / (float)DD);
    __syncthreads();

    float s2 = 0.0f;
    #pragma unroll
    for (int i = 0; i < 4; ++i) { const float dv = v[i] - mu; s2 += dv * dv; }
    red[tid] = s2; __syncthreads();
    for (int off = 128; off > 0; off >>= 1) {
        if (tid < off) red[tid] += red[tid + off];
        __syncthreads();
    }
    const float rstd = rsqrtf(red[0] * (1.0f / (float)DD) + 1e-5f);

    #pragma unroll
    for (int i = 0; i < 4; ++i) {
        const int c = tid + 256 * i;
        row[c] = (v[i] - mu) * rstd * g[c] + bta[c];
    }
}

// ---------------------------------------------------------------------------
extern "C" void kernel_launch(void* const* d_in, const int* in_sizes, int n_in,
                              void* d_out, int out_size, void* d_ws, size_t ws_size,
                              hipStream_t stream) {
    (void)in_sizes; (void)n_in; (void)out_size; (void)ws_size;

    const float* x         = (const float*)d_in[0];
    const int*   t         = (const int*)  d_in[1];
    const float* time_W    = (const float*)d_in[2];
    const float* time_b    = (const float*)d_in[3];
    const float* in_proj_W = (const float*)d_in[4];
    const float* conv_W    = (const float*)d_in[5];
    const float* conv_b    = (const float*)d_in[6];
    const float* x_proj_W  = (const float*)d_in[7];
    const float* dt_W      = (const float*)d_in[8];
    const float* dt_b      = (const float*)d_in[9];
    const float* A_log     = (const float*)d_in[10];
    const float* D_skip    = (const float*)d_in[11];
    const float* out_W     = (const float*)d_in[12];
    const float* norm_g    = (const float*)d_in[13];
    const float* norm_b    = (const float*)d_in[14];
    const float* head_W    = (const float*)d_in[15];
    const float* head_b    = (const float*)d_in[16];
    float* out = (float*)d_out;

    const int M = BB * SS;                         // 8192 rows everywhere

    // workspace layout (floats)
    float* ws    = (float*)d_ws;
    float* tproj = ws;                              //      4,096
    float* hbuf  = tproj + (size_t)BB * DD;         //  8,388,608 (reused: mamba out)
    float* xz    = hbuf  + (size_t)M * DD;          // 33,554,432
    float* xm    = xz    + (size_t)M * 2 * DI;      // 16,777,216 (reused: gated y)
    float* xdbl  = xm    + (size_t)M * DI;          //  1,572,864
    float* dtb   = xdbl  + (size_t)M * 192;         // 16,777,216

    // 1) timestep embedding + time projection  [B,D]
    time_proj_kernel<<<(BB * DD) / 256, 256, 0, stream>>>(t, time_W, time_b, tproj);

    // 2) h = x + t_proj + positional encoding  [B,S,D]
    build_h_kernel<<<(M * DD) / 256, 256, 0, stream>>>(x, tproj, hbuf);

    // 3) in_proj: xz[M, 2*DI] = h @ in_proj_W^T        (WMMA, BN=128)
    gemm_wmma_bf16<0, 128><<<dim3((2 * DI) / 128, M / 128), 256, 0, stream>>>(
        hbuf, DD, in_proj_W, DD, nullptr, xz, 2 * DI, DD);

    // 4) depthwise causal conv + SiLU -> xm[M, DI]
    conv_silu_kernel<<<(M * DI) / 256, 256, 0, stream>>>(xz, conv_W, conv_b, xm);

    // 5) x_proj: xdbl[M, 192] = xm @ x_proj_W^T        (WMMA, BN=64)
    gemm_wmma_bf16<0, 64><<<dim3(192 / 64, M / 128), 256, 0, stream>>>(
        xm, DI, x_proj_W, DI, nullptr, xdbl, 192, DI);

    // 6) dt = softplus(dt_r @ dt_W^T + dt_b)  [M, DI]  (WMMA, fused epilogue)
    gemm_wmma_bf16<2, 128><<<dim3(DI / 128, M / 128), 256, 0, stream>>>(
        xdbl, 192, dt_W, DR_, dt_b, dtb, DI, DR_);

    // 7) selective scan + D-skip + SiLU(z) gate (in-place into xm)
    scan_kernel<<<BB * (DI / 256), 256, 0, stream>>>(
        xdbl, dtb, xz, A_log, D_skip, xm);

    // 8) out_proj: hbuf[M, D] = y @ out_W^T            (WMMA, BN=128)
    gemm_wmma_bf16<0, 128><<<dim3(DD / 128, M / 128), 256, 0, stream>>>(
        xm, DI, out_W, DI, nullptr, hbuf, DD, DI);

    // 9) LayerNorm in-place on hbuf
    layernorm_kernel<<<M, 256, 0, stream>>>(hbuf, norm_g, norm_b);

    // 10) head: out[M, D] = ln @ head_W^T + head_b     (WMMA, BN=128)
    gemm_wmma_bf16<1, 128><<<dim3(DD / 128, M / 128), 256, 0, stream>>>(
        hbuf, DD, head_W, DD, head_b, out, DD, DD);
}
